// Attention_71794673320194
// MI455X (gfx1250) — matrix operand order repaired and checked by reading, not compiled, not run on hardware
//
#include <hip/hip_runtime.h>
#include <hip/hip_bf16.h>
#include <stdint.h>

#define B_   2
#define S_   2048
#define D_   1024
#define H_   16
#define KV_  4
#define HD_  64
#define NREP_ (H_ / KV_)

typedef __attribute__((ext_vector_type(16))) _Float16 v16h;
typedef __attribute__((ext_vector_type(8)))  _Float16 v8h;
typedef __attribute__((ext_vector_type(8)))  float    v8f;

static __device__ __forceinline__ v16h cat8(v8h lo, v8h hi) {
  v16h r;
#pragma unroll
  for (int i = 0; i < 8; ++i) { r[i] = lo[i]; r[i + 8] = hi[i]; }
  return r;
}

static __device__ __forceinline__ v8f wmma_f16(v16h a, v16h b, v8f c) {
  // D = A(16x32 f16) * B(32x16 f16) + C(16x16 f32), wave32
  return __builtin_amdgcn_wmma_f32_16x16x32_f16(
      /*neg_a=*/false, a, /*neg_b=*/false, b,
      /*c_mod=*/(short)0, c, /*reuse_a=*/false, /*reuse_b=*/false);
}

// CDNA5 async global->LDS copy (16B per lane), tracked by ASYNCcnt.
// Generic pointers to __shared__ carry the LDS offset in addr[31:0]
// (flat aperture scheme), which is exactly the VDST operand's meaning.
static __device__ __forceinline__ void async_load_b128(uint32_t lds_addr,
                                                       const void* gaddr) {
  asm volatile("global_load_async_to_lds_b128 %0, %1, off"
               :
               : "v"(lds_addr), "v"((unsigned long long)(uintptr_t)gaddr)
               : "memory");
}
static __device__ __forceinline__ void wait_asynccnt0() {
  asm volatile("s_wait_asynccnt 0x0" ::: "memory");
}

// ---------------------------------------------------------------------------
// Tiled GEMM: C[M,N] = A[M,K](f32 rm) * B[K,N](f32 rm), N/K compile-time.
// Block tile 128x64, 8 waves, each wave 32x32 via 4 WMMA per K-chunk.
// A tile: double-buffered f32 in LDS via GLOBAL_LOAD_ASYNC_TO_LDS_B128,
//         converted to f16 at fragment build (cvt co-executes with WMMA).
// B tile: double-buffered, VGPR-staged transpose to [n][k] f16 so per-lane
//         B fragments are two 8-K runs at ca and ca+16, the same K order as A.
// ---------------------------------------------------------------------------
#define TM   128
#define TN   64
#define TK   32
#define LDAF 36   // f32 words: 144B rows -> 16B aligned, bank-spread
#define LDB  40   // f16 halves: 80B rows -> 16B aligned, bank-spread

template <int N, int K>
__global__ __launch_bounds__(256) void gemm_f32_wmma(
    const float* __restrict__ A, const float* __restrict__ Bm,
    float* __restrict__ C) {
  constexpr int NKCH = K / TK;
  __shared__ float    sA[2][TM * LDAF];
  __shared__ _Float16 sB[2][TN * LDB];
  const int t    = threadIdx.x;
  const int wid  = t >> 5;
  const int lane = t & 31;
  const int m0 = blockIdx.y * TM;
  const int n0 = blockIdx.x * TN;
  const int wm = (wid & 3) * 32;      // 4 wave-rows
  const int wn = (wid >> 2) * 32;     // 2 wave-cols
  const int lrow = lane & 15;
  const int hi   = lane >> 4;         // lane group 0/1
  const int ca   = hi ? 8 : 0;        // A-frag K base (ISA 16-bit A layout)
  const int cb   = hi ? 16 : 0;       // unused; fragB takes its K base from ca

  v8f acc[2][2] = {};
  float bregs[8];

  // --- helpers -------------------------------------------------------------
  auto issueA = [&](int k0, int bsel) {
#pragma unroll
    for (int e = 0; e < 4; ++e) {           // 1024 16B slots / 256 threads
      int slot = e * 256 + t;
      int r  = slot >> 3;                   // tile row (0..127)
      int c4 = (slot & 7) * 4;              // col in floats (0,4,..,28)
      uint32_t la = (uint32_t)(uintptr_t)(const void*)&sA[bsel][r * LDAF + c4];
      async_load_b128(la, &A[(size_t)(m0 + r) * K + (k0 + c4)]);
    }
  };
  auto loadB = [&](int k0) {
#pragma unroll
    for (int e = 0; e < 8; ++e) {
      int idx = e * 256 + t;
      int k = idx >> 6, n = idx & 63;
      bregs[e] = Bm[(size_t)(k0 + k) * N + (n0 + n)];
    }
  };
  auto storeB = [&](int bsel) {
#pragma unroll
    for (int e = 0; e < 8; ++e) {
      int idx = e * 256 + t;
      int k = idx >> 6, n = idx & 63;
      sB[bsel][n * LDB + k] = (_Float16)bregs[e];
    }
  };
  auto fragA = [&](int bsel, int mrow) -> v16h {
    const float* p = &sA[bsel][(mrow + lrow) * LDAF + ca];
    float4 x0 = *(const float4*)(p);        // K = ca .. ca+3
    float4 x1 = *(const float4*)(p + 4);    // K = ca+4 .. ca+7
    float4 x2 = *(const float4*)(p + 16);   // K = 16+ca ..
    float4 x3 = *(const float4*)(p + 20);
    v16h r;
    r[0] = (_Float16)x0.x;  r[1] = (_Float16)x0.y;
    r[2] = (_Float16)x0.z;  r[3] = (_Float16)x0.w;
    r[4] = (_Float16)x1.x;  r[5] = (_Float16)x1.y;
    r[6] = (_Float16)x1.z;  r[7] = (_Float16)x1.w;
    r[8]  = (_Float16)x2.x; r[9]  = (_Float16)x2.y;
    r[10] = (_Float16)x2.z; r[11] = (_Float16)x2.w;
    r[12] = (_Float16)x3.x; r[13] = (_Float16)x3.y;
    r[14] = (_Float16)x3.z; r[15] = (_Float16)x3.w;
    return r;
  };
  auto fragB = [&](int bsel, int ncol) -> v16h {
    const _Float16* p = &sB[bsel][(ncol + lrow) * LDB + ca];
    return cat8(*(const v8h*)p, *(const v8h*)(p + 16));
  };

  // --- pipeline prologue ---------------------------------------------------
  issueA(0, 0);
  loadB(0);
  storeB(0);
  wait_asynccnt0();
  __syncthreads();

  int buf = 0;
  for (int i = 0; i < NKCH; ++i) {
    const bool more = (i + 1) < NKCH;
    if (more) {
      issueA((i + 1) * TK, buf ^ 1);        // async -> other buffer
      loadB((i + 1) * TK);                  // global loads issued early
    }
    v16h af0 = fragA(buf, wm);
    v16h af1 = fragA(buf, wm + 16);
    v16h bf0 = fragB(buf, wn);
    v16h bf1 = fragB(buf, wn + 16);
    acc[0][0] = wmma_f16(af0, bf0, acc[0][0]);
    acc[0][1] = wmma_f16(af0, bf1, acc[0][1]);
    acc[1][0] = wmma_f16(af1, bf0, acc[1][0]);
    acc[1][1] = wmma_f16(af1, bf1, acc[1][1]);
    if (more) {
      storeB(buf ^ 1);
      wait_asynccnt0();
    }
    __syncthreads();
    buf ^= 1;
  }

  // C/D layout: lane holds rows r + 8*hi, col = lane%16
#pragma unroll
  for (int im = 0; im < 2; ++im)
#pragma unroll
    for (int in = 0; in < 2; ++in)
#pragma unroll
      for (int r = 0; r < 8; ++r) {
        int row = m0 + wm + im * 16 + r + hi * 8;
        int col = n0 + wn + in * 16 + lrow;
        C[(size_t)row * N + col] = acc[im][in][r];
      }
}

// ---------------------------------------------------------------------------
// RoPE + pack to f16 in attention-friendly layouts.
//   Qh: [B,H,S,HD]  (A-matrix rows, contiguous HD)   -- scaled by 1/sqrt(HD)
//   Kh: [B,KV,S,HD] (B-matrix of Q*K^T: col=kvpos, contiguous K=HD)
//   Vh: [B,KV,HD,S] (B-matrix of P*V : col=hd, contiguous K=kvpos)
// ---------------------------------------------------------------------------
__global__ __launch_bounds__(256) void rope_pack(
    const float* __restrict__ qf, const float* __restrict__ kf,
    const float* __restrict__ vf, const float* __restrict__ cosT,
    const float* __restrict__ sinT, _Float16* __restrict__ Qh,
    _Float16* __restrict__ Kh, _Float16* __restrict__ Vh) {
  const int tid = blockIdx.x * 256 + threadIdx.x;
  const int nq = B_ * S_ * H_ * HD_;
  if (tid < nq) {
    int d = tid & (HD_ - 1);
    int h = (tid >> 6) & (H_ - 1);
    int s = (tid >> 10) & (S_ - 1);
    int b = tid >> 21;
    size_t rbase = (size_t)(b * S_ + s) * (H_ * HD_) + h * HD_;
    float u  = qf[rbase + d];
    float c  = cosT[s * HD_ + d];
    float sn = sinT[s * HD_ + d];
    float oth = (d < HD_ / 2) ? -qf[rbase + d + HD_ / 2] : qf[rbase + d - HD_ / 2];
    float val = (u * c + oth * sn) * 0.125f;   // fold 1/sqrt(64) into Q
    Qh[((size_t)(b * H_ + h) * S_ + s) * HD_ + d] = (_Float16)val;
  }
  const int nkv = B_ * S_ * KV_ * HD_;
  if (tid < nkv) {
    int d   = tid & (HD_ - 1);
    int kvh = (tid >> 6) & (KV_ - 1);
    int s   = (tid >> 8) & (S_ - 1);
    int b   = tid >> 19;
    size_t rbase = (size_t)(b * S_ + s) * (KV_ * HD_) + kvh * HD_;
    float u  = kf[rbase + d];
    float c  = cosT[s * HD_ + d];
    float sn = sinT[s * HD_ + d];
    float oth = (d < HD_ / 2) ? -kf[rbase + d + HD_ / 2] : kf[rbase + d - HD_ / 2];
    Kh[((size_t)(b * KV_ + kvh) * S_ + s) * HD_ + d] = (_Float16)(u * c + oth * sn);
    Vh[((size_t)(b * KV_ + kvh) * HD_ + d) * S_ + s] = (_Float16)vf[rbase + d];
  }
}

// ---------------------------------------------------------------------------
// Flash attention (causal, GQA). Block = 4 waves x 16 q-rows = 64 q rows.
// Per kv-16 step: 2 WMMA for S = Q*K^T (K=64), online softmax max via
// shfl_xor inside the 16-lane C-layout groups (wave32), row-sum via an
// extra WMMA against an all-ones B tile (o4 = o4*alpha + P@1 tracks the
// softmax denominator with the same rescaling as the output accumulators),
// P relaid via per-wave LDS into A-fragment layout, 4 WMMA for O += P*V.
// ---------------------------------------------------------------------------
__global__ __launch_bounds__(128) void flash_attn(
    const _Float16* __restrict__ Qh, const _Float16* __restrict__ Kh,
    const _Float16* __restrict__ Vh, float* __restrict__ Oa) {
  __shared__ _Float16 sP[4][16 * 16];
  const int wid  = threadIdx.x >> 5;
  const int lane = threadIdx.x & 31;
  const int b = blockIdx.z, h = blockIdx.y;
  const int q0  = blockIdx.x * 64 + wid * 16;
  const int kvh = h / NREP_;
  const _Float16* Qp = Qh + (size_t)(b * H_ + h) * S_ * HD_;
  const _Float16* Kp = Kh + (size_t)(b * KV_ + kvh) * S_ * HD_;
  const _Float16* Vp = Vh + (size_t)(b * KV_ + kvh) * HD_ * S_;
  _Float16* myP = sP[wid];

  const int lrow = lane & 15;
  const int hi   = lane >> 4;
  const int ca   = hi ? 8 : 0;
  const int cb   = hi ? 16 : 0;

  // Q fragments for this wave's 16 rows: K=HD=64 -> two 16x32 A-frags
  v16h qa[2];
  {
    const _Float16* p = Qp + (size_t)(q0 + lrow) * HD_;
#pragma unroll
    for (int tk = 0; tk < 2; ++tk) {
      const _Float16* pp = p + tk * 32 + ca;
      qa[tk] = cat8(*(const v8h*)pp, *(const v8h*)(pp + 16));
    }
  }

  // All-ones B tile restricted to the valid K rows (0..15 -> lanes 0-15):
  // o4 = sum over kv of P -> softmax denominator, auto-rescaled by alpha.
  v16h onesb;
#pragma unroll
  for (int i = 0; i < 16; ++i)
    onesb[i] = hi ? (_Float16)0.0f : (_Float16)1.0f;

  v8f o[5] = {};                       // o[0..3] = output, o[4] = row sums
  float mstat[8];
#pragma unroll
  for (int r = 0; r < 8; ++r) mstat[r] = -1e30f;

  for (int kv0 = 0; kv0 <= q0; kv0 += 16) {   // causal: tiles fully <= diag
    // ---- S tile = Q * K^T
    v8f s = {};
#pragma unroll
    for (int tk = 0; tk < 2; ++tk) {
      const _Float16* pp = Kp + (size_t)(kv0 + lrow) * HD_ + tk * 32 + ca;
      v16h kb = cat8(*(const v8h*)pp, *(const v8h*)(pp + 16));
      s = wmma_f16(qa[tk], kb, s);
    }
    // ---- diagonal-tile causal mask
    if (kv0 == q0) {
#pragma unroll
      for (int r = 0; r < 8; ++r)
        if (lrow > r + hi * 8) s[r] = -1e30f;
    }
    // ---- online softmax: row max across the 16-lane group, then rescale
    float p[8], alpha[8];
#pragma unroll
    for (int r = 0; r < 8; ++r) {
      float v = s[r];
      v = fmaxf(v, __shfl_xor(v, 1, 32));
      v = fmaxf(v, __shfl_xor(v, 2, 32));
      v = fmaxf(v, __shfl_xor(v, 4, 32));
      v = fmaxf(v, __shfl_xor(v, 8, 32));
      float mn = fmaxf(mstat[r], v);
      alpha[r] = __expf(mstat[r] - mn);
      mstat[r] = mn;
      p[r] = __expf(s[r] - mn);
    }
#pragma unroll
    for (int c = 0; c < 5; ++c)
#pragma unroll
      for (int r = 0; r < 8; ++r) o[c][r] *= alpha[r];

    // ---- P (C layout) -> LDS -> A-fragment layout (K 16..31 zero)
#pragma unroll
    for (int r = 0; r < 8; ++r)
      myP[(r + hi * 8) * 16 + lrow] = (_Float16)p[r];
    v16h pa;
    {
      v8h plo = *(const v8h*)&myP[lrow * 16];
      v8h phi = *(const v8h*)&myP[lrow * 16 + 8];
#pragma unroll
      for (int i = 0; i < 8; ++i) { pa[i] = hi ? (_Float16)0.0f : plo[i]; pa[i + 8] = hi ? (_Float16)0.0f : phi[i]; }
    }
    // ---- O += P * V  (V B-frags: only lanes 0-15 carry valid K=0..15)
#pragma unroll
    for (int c = 0; c < 4; ++c) {
      v16h vb = {};
      if (!hi) {
        const _Float16* pp = Vp + (size_t)(c * 16 + lrow) * S_ + kv0;
        vb = cat8(*(const v8h*)pp, *(const v8h*)(pp + 8));
      }
      o[c] = wmma_f16(pa, vb, o[c]);
    }
    // ---- denominator: o4 += P @ ones (every column = rowsum)
    o[4] = wmma_f16(pa, onesb, o[4]);
  }

  // ---- finalize: O / l -> attention buffer [B*S][H*HD] (f32, GEMM A input)
#pragma unroll
  for (int c = 0; c < 4; ++c)
#pragma unroll
    for (int r = 0; r < 8; ++r) {
      int row = q0 + r + hi * 8;
      int col = h * HD_ + c * 16 + lrow;
      Oa[(size_t)(b * S_ + row) * (H_ * HD_) + col] = o[c][r] / o[4][r];
    }
}

// ---------------------------------------------------------------------------
extern "C" void kernel_launch(void* const* d_in, const int* in_sizes, int n_in,
                              void* d_out, int out_size, void* d_ws, size_t ws_size,
                              hipStream_t stream) {
  const float* x    = (const float*)d_in[0];
  const float* cosT = (const float*)d_in[1];
  const float* sinT = (const float*)d_in[2];
  const float* Wq   = (const float*)d_in[3];
  const float* Wk   = (const float*)d_in[4];
  const float* Wv   = (const float*)d_in[5];
  const float* Wo   = (const float*)d_in[6];
  float* out = (float*)d_out;

  const size_t M = (size_t)B_ * S_;                 // 4096 rows
  char* w = (char*)d_ws;
  float*    qf = (float*)w;    w += M * (H_ * HD_)  * sizeof(float);
  float*    kf = (float*)w;    w += M * (KV_ * HD_) * sizeof(float);
  float*    vf = (float*)w;    w += M * (KV_ * HD_) * sizeof(float);
  _Float16* Qh = (_Float16*)w; w += M * (H_ * HD_)  * sizeof(_Float16);
  _Float16* Kh = (_Float16*)w; w += M * (KV_ * HD_) * sizeof(_Float16);
  _Float16* Vh = (_Float16*)w; w += M * (KV_ * HD_) * sizeof(_Float16);
  float*  attn = (float*)w;    w += M * (H_ * HD_)  * sizeof(float);
  (void)ws_size; (void)in_sizes; (void)n_in; (void)out_size;

  // 1) QKV projections (f16 WMMA, f32 accumulate, async-LDS A staging)
  gemm_f32_wmma<H_ * HD_, D_><<<dim3((H_ * HD_) / TN, M / TM), 256, 0, stream>>>(
      x, Wq, qf);
  gemm_f32_wmma<KV_ * HD_, D_><<<dim3((KV_ * HD_) / TN, M / TM), 256, 0, stream>>>(
      x, Wk, kf);
  gemm_f32_wmma<KV_ * HD_, D_><<<dim3((KV_ * HD_) / TN, M / TM), 256, 0, stream>>>(
      x, Wv, vf);

  // 2) RoPE + pack to f16 fragment-friendly layouts
  const int nq = B_ * S_ * H_ * HD_;
  rope_pack<<<(nq + 255) / 256, 256, 0, stream>>>(qf, kf, vf, cosT, sinT,
                                                  Qh, Kh, Vh);

  // 3) causal flash attention (bulk of the FLOPs)
  flash_attn<<<dim3(S_ / 64, H_, B_), 128, 0, stream>>>(Qh, Kh, Vh, attn);

  // 4) output projection
  gemm_f32_wmma<D_, D_><<<dim3(D_ / TN, M / TM), 256, 0, stream>>>(
      attn, Wo, out);
}
